// Text2Motion_RNN_15083925144392
// MI455X (gfx1250) — matrix operand order repaired and checked by reading, not compiled, not run on hardware
//
#include <hip/hip_runtime.h>

// ---------------- problem constants ----------------
#define Bsz   4
#define Tlen  2048
#define Dm    512
#define Hf    2048
#define Lnum  12
#define Vocab 1025
#define VPAD  1088          // head N padded to multiple of 64 (4-tile groups)
#define RING  8             // inter-layer ring-buffer depth (power of 2)
#define DD    (512*512)
#define LAYER_ELEMS (5*DD + 2*512*2048)   // Wr,Wk,Wv,Wo,Wr_ffn (512x512) + Wk_ffn(512x2048) + Wv_ffn(2048x512)
#define CHUNK (Lnum*Bsz*Dm)               // one state array: [12][4][512] floats

typedef __attribute__((ext_vector_type(16))) __bf16 v16bf;
typedef __attribute__((ext_vector_type(8)))  float  v8f;

// =====================================================================
// weight convert+transpose: dst[k*Npad + n] = bf16(src[n*K + k]); pad with 0
// =====================================================================
__global__ void k_transpose_bf16(const float* __restrict__ src, __bf16* __restrict__ dst,
                                 int N, int K, int Npad) {
  int idx = blockIdx.x * blockDim.x + threadIdx.x;
  int total = K * Npad;
  if (idx >= total) return;
  int k = idx / Npad;
  int n = idx - k * Npad;
  dst[idx] = (n < N) ? (__bf16)src[(size_t)n * K + k] : (__bf16)0.0f;
}

// =====================================================================
// state + flag init (runs every launch; deterministic)
// state layout: [xx_att, aa, bb, pp, xx_ffn] each [12][4][512] f32
// =====================================================================
__global__ void k_init(float* states, int nstate, int* flags, int nflags) {
  int i = blockIdx.x * blockDim.x + threadIdx.x;
  if (i < nstate) {
    int which = i / CHUNK;                      // 0..4
    states[i] = (which == 3) ? -1e30f : 0.0f;   // pp starts at -1e30
  }
  if (i < nflags) flags[i] = 0;
}

// =====================================================================
// WMMA fragment loaders (CDNA5 16-bit layouts, ISA 7.12.2)
// A (16x32, M x K): lane L carries M=L&15, K-base=(L>>4)*8; elems 0..7 ->
//   K=base..base+7, elems 8..15 -> K=base+16..base+23.
//   Only D rows 0..3 are ever read back, and D rows depend only on the same
//   A row -> rows 4..15 may hold ANY data. All lanes read row (lane & 3):
//   branch-free, no zero-fill, no EXEC manipulation.
// B (32x16, K x N): lane L holds N=n0+(L&15), K-base=(L>>4)*8; elems 0..7 ->
//   K=base..base+7, elems 8..15 -> K=base+16..base+23 (Wt[k][n], stride Npad).
// =====================================================================
__device__ __forceinline__ v16bf load_A_frag(const __bf16* Ab, int stride, int k0, int lane) {
  int M  = lane & 3;
  int kb = k0 + ((lane >> 4) << 3);
  const __bf16* p = Ab + M * stride + kb;
  v16bf a;
#pragma unroll
  for (int j = 0; j < 8; ++j)  a[j]     = p[j];
#pragma unroll
  for (int j = 0; j < 8; ++j)  a[8 + j] = p[16 + j];
  return a;
}

__device__ __forceinline__ v16bf load_B_frag(const __bf16* Wt, int Npad, int k0, int n0, int lane) {
  const __bf16* p = Wt + (size_t)(k0 + ((lane >> 4) << 3)) * Npad + n0 + (lane & 15);
  v16bf b;
#pragma unroll
  for (int j = 0; j < 8; ++j) { b[j] = p[(size_t)j * Npad]; b[8 + j] = p[(size_t)(16 + j) * Npad]; }
  return b;
}

// 4 independent 16x16 output tiles (64 contiguous N columns) per wave:
// one A-fragment LDS load feeds 4 WMMAs with independent accumulators
// (no D->C RAW hazard NOPs, 8 B-loads in flight per wait group).
__device__ __forceinline__ void gemm_tile4(const __bf16* Ab, int K, const __bf16* Wt, int Npad,
                                           int n0, int lane, v8f acc[4]) {
  // warm the first weight rows of this 64-column slab (-> global_prefetch_b8)
  __builtin_prefetch(Wt + (size_t)(lane + 32) * Npad + n0, 0, 1);
  for (int k0 = 0; k0 < K; k0 += 32) {
    v16bf a = load_A_frag(Ab, K, k0, lane);
#pragma unroll
    for (int u = 0; u < 4; ++u) {
      v16bf b = load_B_frag(Wt, Npad, k0, n0 + u * 16, lane);
      acc[u] = __builtin_amdgcn_wmma_f32_16x16x32_bf16(false, a, false, b, (short)0, acc[u],
                                                       false, false);
    }
  }
}

// D-matrix layout: lanes 0..15 carry M=0..7 in acc[0..7]; only M=0..3 valid.
// mode 0: store  | 1: sigmoid store | 3: dst += x | 4: dst += mul * x
__device__ void gemm_rows4_f32(const __bf16* Ab, int K, const __bf16* Wt, int Npad, int N,
                               float* dst, const float* mul, int mode, int tid) {
  int wave = tid >> 5, lane = tid & 31;
  for (int n0 = wave * 64; n0 < N; n0 += 8 * 64) {
    v8f z = {};
    v8f acc[4] = {z, z, z, z};
    gemm_tile4(Ab, K, Wt, Npad, n0, lane, acc);
    if (lane < 16) {
#pragma unroll
      for (int u = 0; u < 4; ++u) {
        int n = n0 + u * 16 + lane;
#pragma unroll
        for (int m = 0; m < 4; ++m) {
          float x = acc[u][m];
          if (mode == 1)       dst[m * N + n]  = 1.0f / (1.0f + __expf(-x));
          else if (mode == 0)  dst[m * N + n]  = x;
          else if (mode == 3)  dst[m * N + n] += x;
          else                 dst[m * N + n] += mul[m * N + n] * x;
        }
      }
    }
  }
}

// FFN k path: relu(x)^2, stored as bf16 A-operand for the second FFN GEMM
__device__ void gemm_rows4_relusq(const __bf16* Ab, int K, const __bf16* Wt, int Npad, int N,
                                  __bf16* dstb, int tid) {
  int wave = tid >> 5, lane = tid & 31;
  for (int n0 = wave * 64; n0 < N; n0 += 8 * 64) {
    v8f z = {};
    v8f acc[4] = {z, z, z, z};
    gemm_tile4(Ab, K, Wt, Npad, n0, lane, acc);
    if (lane < 16) {
#pragma unroll
      for (int u = 0; u < 4; ++u) {
        int n = n0 + u * 16 + lane;
#pragma unroll
        for (int m = 0; m < 4; ++m) {
          float x = fmaxf(acc[u][m], 0.0f);
          dstb[m * N + n] = (__bf16)(x * x);
        }
      }
    }
  }
}

// head projection: N = 1025 (padded to 1088), output f32 logits [B][T][V]
__device__ void gemm_head(const __bf16* Ab, const __bf16* Wt, int t, float* out, int tid) {
  int wave = tid >> 5, lane = tid & 31;
  for (int n0 = wave * 64; n0 < VPAD; n0 += 8 * 64) {
    v8f z = {};
    v8f acc[4] = {z, z, z, z};
    gemm_tile4(Ab, Dm, Wt, VPAD, n0, lane, acc);
    if (lane < 16) {
#pragma unroll
      for (int u = 0; u < 4; ++u) {
        int n = n0 + u * 16 + lane;
        if (n < Vocab) {
#pragma unroll
          for (int m = 0; m < 4; ++m)
            out[((size_t)m * Tlen + t) * Vocab + n] = acc[u][m];
        }
      }
    }
  }
}

// =====================================================================
// elementwise helpers (4x512 = 2048 elems over 256 threads, 8 each)
// =====================================================================
__device__ void layernorm4(const float* in, float* out, const float* w, const float* bvec,
                           float* red, int tid) {
  __syncthreads();
  if (tid < 8) red[tid] = 0.0f;
  __syncthreads();
  int row = tid >> 6;
  int c0  = (tid & 63) * 8;
  float v[8], s = 0.0f, s2 = 0.0f;
#pragma unroll
  for (int j = 0; j < 8; ++j) { v[j] = in[row * Dm + c0 + j]; s += v[j]; s2 += v[j] * v[j]; }
  atomicAdd(&red[row * 2 + 0], s);
  atomicAdd(&red[row * 2 + 1], s2);
  __syncthreads();
  float mean = red[row * 2 + 0] * (1.0f / Dm);
  float var  = red[row * 2 + 1] * (1.0f / Dm) - mean * mean;
  float inv  = rsqrtf(var + 1e-5f);
#pragma unroll
  for (int j = 0; j < 8; ++j)
    out[row * Dm + c0 + j] = (v[j] - mean) * inv * w[c0 + j] + bvec[c0 + j];
}

__device__ void build_mix_A(const float* xs, const float* prev, const float* tmix,
                            __bf16* Ab, int tid) {
#pragma unroll
  for (int j = 0; j < 8; ++j) {
    int i = tid * 8 + j;           // [4][512] flat
    int c = i & 511;
    float mw = tmix[c];
    Ab[i] = (__bf16)(xs[i] * mw + prev[i] * (1.0f - mw));
  }
}

__device__ void copy2048(const float* src, float* dst, int tid) {
#pragma unroll
  for (int j = 0; j < 8; ++j) { int i = tid * 8 + j; dst[i] = src[i]; }
}

__device__ void wkv_update(const float* k, const float* v, const float* r,
                           float* aa, float* bb, float* pp,
                           const float* tf, const float* td, __bf16* Ab, int tid) {
#pragma unroll
  for (int j = 0; j < 8; ++j) {
    int i = tid * 8 + j;
    int c = i & 511;
    float kk = k[i], vv = v[i];
    float paa = aa[i], pbb = bb[i], ppp = pp[i];
    float ww = tf[c] + kk;
    float q  = fmaxf(ppp, ww);
    float e1 = __expf(ppp - q), e2 = __expf(ww - q);
    float a  = e1 * paa + e2 * vv;
    float b  = e1 * pbb + e2;
    float rr = 1.0f / (1.0f + __expf(-r[i]));
    Ab[i] = (__bf16)(rr * a / b);
    float ww2 = ppp + td[c];
    float q2  = fmaxf(ww2, kk);
    float e1b = __expf(ww2 - q2), e2b = __expf(kk - q2);
    aa[i] = e1b * paa + e2b * vv;
    bb[i] = e1b * pbb + e2b;
    pp[i] = q2;
  }
}

// =====================================================================
// persistent layer-pipelined kernel: blocks 0..11 = layers, block 12 = head
// =====================================================================
__global__ __launch_bounds__(256, 1)
void k_rwkv_pipeline(const float* __restrict__ x,
                     const float* ln0_w, const float* ln0_b,
                     const float* ln1_w, const float* ln1_b,
                     const float* ln2_w, const float* ln2_b,
                     const float* tmk, const float* tmv, const float* tmr,
                     const float* tfirst, const float* tdecay,
                     const float* fmk, const float* fmr,
                     const float* lnf_w, const float* lnf_b,
                     const __bf16* __restrict__ Wbf,
                     float* states, float* Hbuf, int* prod, int* cons,
                     float* __restrict__ out) {
  __shared__ float  sh_h[2048];        // residual h  [4][512]
  __shared__ float  sh_xs[2048];       // LN output   [4][512]
  __shared__ float  sh_kv[4096];       // k|v [4][512] each; reused as kf bf16 [4][2048]
  __shared__ float  sh_r[2048];        // r / rf
  __shared__ __bf16 sh_A[8192];        // A operand, up to [4][2048]
  __shared__ float  sh_red[8];

  const int blk = blockIdx.x, tid = threadIdx.x;

  if (blk < Lnum) {
    const int l = blk;
    const __bf16* Wl  = Wbf + (size_t)l * LAYER_ELEMS;
    const __bf16* Wr  = Wl;
    const __bf16* Wk  = Wl + 1 * DD;
    const __bf16* Wv  = Wl + 2 * DD;
    const __bf16* Wo  = Wl + 3 * DD;
    const __bf16* Wrf = Wl + 4 * DD;
    const __bf16* Wkf = Wl + 5 * DD;                 // [512][2048]
    const __bf16* Wvf = Wl + 5 * DD + 512 * 2048;    // [2048][512]
    float* xx_att = states + 0 * CHUNK + l * 2048;
    float* aa     = states + 1 * CHUNK + l * 2048;
    float* bb     = states + 2 * CHUNK + l * 2048;
    float* pp     = states + 3 * CHUNK + l * 2048;
    float* xx_ffn = states + 4 * CHUNK + l * 2048;

    for (int t = 0; t < Tlen; ++t) {
      // -------- acquire input h(t) --------
      if (l == 0) {
#pragma unroll
        for (int j = 0; j < 8; ++j) {
          int i = tid * 8 + j; int b = i >> 9, c = i & 511;
          sh_xs[i] = x[((size_t)b * Tlen + t) * Dm + c];
        }
        layernorm4(sh_xs, sh_h, ln0_w, ln0_b, sh_red, tid);
        __syncthreads();
      } else {
        if (tid == 0) {
          while (__hip_atomic_load(&prod[l - 1], __ATOMIC_ACQUIRE, __HIP_MEMORY_SCOPE_AGENT) <= t)
            __builtin_amdgcn_s_sleep(1);
        }
        __syncthreads();
        const float* src = Hbuf + ((size_t)(l - 1) * RING + (t & (RING - 1))) * 2048;
        copy2048(src, sh_h, tid);
        __syncthreads();
        if (tid == 0)
          __hip_atomic_store(&cons[l - 1], t + 1, __ATOMIC_RELEASE, __HIP_MEMORY_SCOPE_AGENT);
      }

      // -------- attention (WKV) --------
      layernorm4(sh_h, sh_xs, ln1_w + l * Dm, ln1_b + l * Dm, sh_red, tid);
      __syncthreads();
      build_mix_A(sh_xs, xx_att, tmk + l * Dm, sh_A, tid);   __syncthreads();
      gemm_rows4_f32(sh_A, Dm, Wk, Dm, Dm, sh_kv, nullptr, 0, tid);          __syncthreads();
      build_mix_A(sh_xs, xx_att, tmv + l * Dm, sh_A, tid);   __syncthreads();
      gemm_rows4_f32(sh_A, Dm, Wv, Dm, Dm, sh_kv + 2048, nullptr, 0, tid);   __syncthreads();
      build_mix_A(sh_xs, xx_att, tmr + l * Dm, sh_A, tid);   __syncthreads();
      gemm_rows4_f32(sh_A, Dm, Wr, Dm, Dm, sh_r, nullptr, 0, tid);           __syncthreads();
      copy2048(sh_xs, xx_att, tid);           // new xx_att = ln1 output
      wkv_update(sh_kv, sh_kv + 2048, sh_r, aa, bb, pp,
                 tfirst + l * Dm, tdecay + l * Dm, sh_A, tid);
      __syncthreads();
      gemm_rows4_f32(sh_A, Dm, Wo, Dm, Dm, sh_h, nullptr, 3, tid);           __syncthreads();

      // -------- FFN --------
      layernorm4(sh_h, sh_xs, ln2_w + l * Dm, ln2_b + l * Dm, sh_red, tid);
      __syncthreads();
      build_mix_A(sh_xs, xx_ffn, fmr + l * Dm, sh_A, tid);   __syncthreads();
      gemm_rows4_f32(sh_A, Dm, Wrf, Dm, Dm, sh_r, nullptr, 1, tid);          __syncthreads();
      build_mix_A(sh_xs, xx_ffn, fmk + l * Dm, sh_A, tid);   __syncthreads();
      copy2048(sh_xs, xx_ffn, tid);           // new xx_ffn = ln2 output
      __bf16* kf = (__bf16*)sh_kv;            // [4][2048] bf16 reuse
      gemm_rows4_relusq(sh_A, Dm, Wkf, Hf, Hf, kf, tid);                     __syncthreads();
      gemm_rows4_f32(kf, Hf, Wvf, Dm, Dm, sh_h, sh_r, 4, tid);               __syncthreads();

      // -------- publish h(t) --------
      if (tid == 0) {
        while (__hip_atomic_load(&cons[l], __ATOMIC_ACQUIRE, __HIP_MEMORY_SCOPE_AGENT) <= t - RING)
          __builtin_amdgcn_s_sleep(1);
      }
      __syncthreads();
      float* dstH = Hbuf + ((size_t)l * RING + (t & (RING - 1))) * 2048;
      copy2048(sh_h, dstH, tid);
      __syncthreads();
      if (tid == 0)
        __hip_atomic_store(&prod[l], t + 1, __ATOMIC_RELEASE, __HIP_MEMORY_SCOPE_AGENT);
    }
  } else {
    // -------- head block: lnf + vocab projection --------
    const __bf16* Whead = Wbf + (size_t)Lnum * LAYER_ELEMS;   // [512][1088]
    for (int t = 0; t < Tlen; ++t) {
      if (tid == 0) {
        while (__hip_atomic_load(&prod[Lnum - 1], __ATOMIC_ACQUIRE, __HIP_MEMORY_SCOPE_AGENT) <= t)
          __builtin_amdgcn_s_sleep(1);
      }
      __syncthreads();
      const float* src = Hbuf + ((size_t)(Lnum - 1) * RING + (t & (RING - 1))) * 2048;
      copy2048(src, sh_h, tid);
      __syncthreads();
      if (tid == 0)
        __hip_atomic_store(&cons[Lnum - 1], t + 1, __ATOMIC_RELEASE, __HIP_MEMORY_SCOPE_AGENT);
      layernorm4(sh_h, sh_xs, lnf_w, lnf_b, sh_red, tid);
      __syncthreads();
#pragma unroll
      for (int j = 0; j < 8; ++j) { int i = tid * 8 + j; sh_A[i] = (__bf16)sh_xs[i]; }
      __syncthreads();
      gemm_head(sh_A, Whead, t, out, tid);
      __syncthreads();
    }
  }
}

// =====================================================================
// host side
// =====================================================================
extern "C" void kernel_launch(void* const* d_in, const int* in_sizes, int n_in,
                              void* d_out, int out_size, void* d_ws, size_t ws_size,
                              hipStream_t stream) {
  const float* x      = (const float*)d_in[0];
  const float* ln0_w  = (const float*)d_in[1];
  const float* ln0_b  = (const float*)d_in[2];
  const float* ln1_w  = (const float*)d_in[3];
  const float* ln1_b  = (const float*)d_in[4];
  const float* ln2_w  = (const float*)d_in[5];
  const float* ln2_b  = (const float*)d_in[6];
  const float* tmk    = (const float*)d_in[7];
  const float* tmv    = (const float*)d_in[8];
  const float* tmr    = (const float*)d_in[9];
  const float* tfirst = (const float*)d_in[10];
  const float* tdecay = (const float*)d_in[11];
  const float* att_Wr = (const float*)d_in[12];
  const float* att_Wk = (const float*)d_in[13];
  const float* att_Wv = (const float*)d_in[14];
  const float* att_Wo = (const float*)d_in[15];
  const float* fmk    = (const float*)d_in[16];
  const float* fmr    = (const float*)d_in[17];
  const float* ffn_Wk = (const float*)d_in[18];
  const float* ffn_Wv = (const float*)d_in[19];
  const float* ffn_Wr = (const float*)d_in[20];
  const float* lnf_w  = (const float*)d_in[21];
  const float* lnf_b  = (const float*)d_in[22];
  const float* head_W = (const float*)d_in[23];
  float* out = (float*)d_out;

  // workspace carve-up: bf16 weights | states | ring buffers | flags  (~84 MB)
  const size_t WB_ELEMS = (size_t)Lnum * LAYER_ELEMS + (size_t)Dm * VPAD;
  __bf16* Wbf   = (__bf16*)d_ws;
  float* states = (float*)((char*)d_ws + WB_ELEMS * sizeof(__bf16));
  float* Hbuf   = states + 5 * CHUNK;
  int*   flags  = (int*)(Hbuf + (size_t)Lnum * RING * 2048);
  int*   prod   = flags;
  int*   cons   = flags + 16;

  // 1) init states + flags
  {
    int nstate = 5 * CHUNK;
    k_init<<<(nstate + 255) / 256, 256, 0, stream>>>(states, nstate, flags, 32);
  }

  // 2) convert + transpose all weights to L2-resident bf16
  auto T2 = [&](const float* src, size_t srcOff, size_t dstOff, int N, int K, int Npad) {
    int total = K * Npad;
    k_transpose_bf16<<<(total + 255) / 256, 256, 0, stream>>>(src + srcOff, Wbf + dstOff, N, K, Npad);
  };
  for (int l = 0; l < Lnum; ++l) {
    size_t base = (size_t)l * LAYER_ELEMS;
    T2(att_Wr, (size_t)l * DD,          base + 0 * DD, Dm, Dm, Dm);
    T2(att_Wk, (size_t)l * DD,          base + 1 * DD, Dm, Dm, Dm);
    T2(att_Wv, (size_t)l * DD,          base + 2 * DD, Dm, Dm, Dm);
    T2(att_Wo, (size_t)l * DD,          base + 3 * DD, Dm, Dm, Dm);
    T2(ffn_Wr, (size_t)l * DD,          base + 4 * DD, Dm, Dm, Dm);
    T2(ffn_Wk, (size_t)l * Hf * Dm,     base + 5 * DD, Hf, Dm, Hf);                   // [H][D] -> [D][H]
    T2(ffn_Wv, (size_t)l * Dm * Hf,     base + 5 * DD + (size_t)Dm * Hf, Dm, Hf, Dm); // [D][H] -> [H][D]
  }
  T2(head_W, 0, (size_t)Lnum * LAYER_ELEMS, Vocab, Dm, VPAD);

  // 3) persistent layer-pipelined forward (12 layer blocks + 1 head block)
  k_rwkv_pipeline<<<Lnum + 1, 256, 0, stream>>>(
      x, ln0_w, ln0_b, ln1_w, ln1_b, ln2_w, ln2_b,
      tmk, tmv, tmr, tfirst, tdecay, fmk, fmr, lnf_w, lnf_b,
      Wbf, states, Hbuf, prod, cons, out);

  (void)in_sizes; (void)n_in; (void)out_size; (void)ws_size;
}
